// Block_21801253994551
// MI455X (gfx1250) — matrix operand order repaired and checked by reading, not compiled, not run on hardware
//
#include <hip/hip_runtime.h>
#include <cstddef>
#include <cstdint>

// ---------------------------------------------------------------------------
// RWKV-7 block for MI455X (gfx1250): bf16 WMMA GEMMs + fused epilogues +
// async global->LDS staging + sequential per-head state scan.
// B=4, T=4096, C=1024, H=16, N=64.
// ---------------------------------------------------------------------------

#define RB 4
#define RT 4096
#define RC 1024
#define RH 16
#define RN 64
#define RM (RB * RT)          // 16384 rows
#define BTC ((size_t)RM * RC) // 16,777,216 elements

#define USE_ASYNC_LDS 1       // gfx1250 GLOBAL_LOAD_ASYNC_TO_LDS path

typedef __bf16 bf16;
typedef __attribute__((ext_vector_type(16))) __bf16 v16bf;
typedef __attribute__((ext_vector_type(8)))  __bf16 v8bf;
typedef __attribute__((ext_vector_type(8)))  float  v8f;

__device__ __forceinline__ bf16 f2bf(float f) {
  unsigned u = __builtin_bit_cast(unsigned, f);
  u = (u + 0x7FFFu + ((u >> 16) & 1u)) >> 16;
  return __builtin_bit_cast(bf16, (unsigned short)u);
}
__device__ __forceinline__ float bf2f(bf16 b) {
  unsigned u = ((unsigned)__builtin_bit_cast(unsigned short, b)) << 16;
  return __builtin_bit_cast(float, u);
}
__device__ __forceinline__ float sigm(float x) { return 1.0f / (1.0f + expf(-x)); }

// ---------------------------------------------------------------------------
// WMMA GEMM:  C[M,N] = A[M,K] (bf16) @ B[K,N] (bf16)
//   AMODE 0: A used as-is (async global->LDS staging, ASYNCcnt tracked)
//   AMODE 1: A is ln(x) bf16; operand = a + (a[t-1]-a)*mix[k]  (RWKV shift,
//            T == 4096 is a power of two -> mask instead of modulo)
//   EMODE:   0 f32 store | 1 f32 accumulate | 2 bf16 | 3 tanh->bf16
//            4 sigmoid->bf16 | 5 relu^2 ->bf16
// Tile: 128x128 per block, 256 threads (8 waves, 4x2), K step 32.
// Each wave: 32x64 = 2x4 WMMA accumulators; B-frags cached in registers.
// ---------------------------------------------------------------------------
#define GBM 128
#define GBN 128
#define GBK 32
#define LDA 40   // padded LDS row stride (bf16): 80 bytes, keeps 16B alignment

enum { EP_F32 = 0, EP_ADDF32 = 1, EP_BF16 = 2, EP_TANH = 3, EP_SIG = 4, EP_RELUSQ = 5 };

template <int AMODE, int EMODE>
__global__ __launch_bounds__(256) void gemm_wmma_kernel(
    const bf16* __restrict__ A, const float* __restrict__ mix,
    const bf16* __restrict__ Bw, void* __restrict__ Cout,
    int M, int N, int K)
{
  __shared__ bf16 As[GBM][LDA];
  __shared__ bf16 Bs[GBN][LDA];   // transposed: Bs[n][k]

  const int tid  = threadIdx.x;
  const int m0   = blockIdx.y * GBM;
  const int n0   = blockIdx.x * GBN;
  const int wave = tid >> 5;
  const int lane = tid & 31;
  const int lr   = lane & 15;
  const int hi   = lane >> 4;
  const int wr   = (wave >> 1) * 32;   // wave row offset in tile (0/32/64/96)
  const int wc   = (wave & 1) * 64;    // wave col offset in tile (0/64)

  v8f acc[2][4] = {};

  for (int k0 = 0; k0 < K; k0 += GBK) {
    // ================= stage A tile: 128 rows x 32 cols =================
    if (AMODE == 0) {
#if USE_ASYNC_LDS
      // gfx1250 async copy: memory -> LDS without VGPR round trip.
      #pragma unroll
      for (int it = 0; it < 2; ++it) {
        int idx = tid + it * 256;        // 0..511
        int row = idx >> 2;
        int seg = (idx & 3) * 8;         // 8-col (16B) segment
        int gm  = m0 + row;              // M is always a multiple of 128
        int gk  = k0 + seg;
        const bf16* gp = &A[(size_t)gm * K + gk];
        unsigned lds = (unsigned)(uintptr_t)&As[row][seg];
        asm volatile("global_load_async_to_lds_b128 %0, %1, off"
                     :: "v"(lds), "v"(gp) : "memory");
      }
#else
      v8bf av[2];
      #pragma unroll
      for (int it = 0; it < 2; ++it) {
        int idx = tid + it * 256;
        int row = idx >> 2, seg = (idx & 3) * 8;
        av[it] = *(const v8bf*)&A[(size_t)(m0 + row) * K + k0 + seg];
      }
      #pragma unroll
      for (int it = 0; it < 2; ++it) {
        int idx = tid + it * 256;
        int row = idx >> 2, seg = (idx & 3) * 8;
        *(v8bf*)&As[row][seg] = av[it];
      }
#endif
      if (k0 + GBK < K)
        __builtin_prefetch(&A[(size_t)(m0 + (tid >> 1)) * K + k0 + GBK], 0, 1);
    } else {
      // token-shift mix fused at LDS-fill time; gather first, store second
      v8bf cur[2], prv[2];
      bool hasp[2];
      #pragma unroll
      for (int it = 0; it < 2; ++it) {
        int idx = tid + it * 256;
        int row = idx >> 2, seg = (idx & 3) * 8;
        int gm  = m0 + row;
        int gk  = k0 + seg;
        cur[it] = *(const v8bf*)&A[(size_t)gm * K + gk];
        hasp[it] = (gm & (RT - 1)) != 0;
        prv[it] = hasp[it] ? *(const v8bf*)&A[(size_t)(gm - 1) * K + gk] : v8bf{};
      }
      #pragma unroll
      for (int it = 0; it < 2; ++it) {
        int idx = tid + it * 256;
        int row = idx >> 2, seg = (idx & 3) * 8;
        int gk  = k0 + seg;
        v8bf outv;
        #pragma unroll
        for (int e = 0; e < 8; ++e) {
          float c = bf2f(cur[it][e]);
          float p = hasp[it] ? bf2f(prv[it][e]) : 0.0f;
          outv[e] = f2bf(c + (p - c) * mix[gk + e]);
        }
        *(v8bf*)&As[row][seg] = outv;
      }
      if (k0 + GBK < K)
        __builtin_prefetch(&A[(size_t)(m0 + (tid >> 1)) * K + k0 + GBK], 0, 1);
    }

    // ========== stage B tile transposed: 32 k-rows x 128 n ==========
    {
      v8bf bv[2];
      #pragma unroll
      for (int it = 0; it < 2; ++it) {
        int idx = tid + it * 256;        // 0..511
        int kk  = idx >> 4;              // 0..31
        int seg = (idx & 15) * 8;        // n segment
        int gn  = n0 + seg;
        if (gn + 8 <= N) {
          bv[it] = *(const v8bf*)&Bw[(size_t)(k0 + kk) * N + gn];
        } else {
          #pragma unroll
          for (int e = 0; e < 8; ++e)
            bv[it][e] = (gn + e < N) ? Bw[(size_t)(k0 + kk) * N + gn + e] : f2bf(0.0f);
        }
      }
      #pragma unroll
      for (int it = 0; it < 2; ++it) {
        int idx = tid + it * 256;
        int kk  = idx >> 4;
        int seg = (idx & 15) * 8;
        #pragma unroll
        for (int e = 0; e < 8; ++e) Bs[seg + e][kk] = bv[it][e];
      }
    }

#if USE_ASYNC_LDS
    if (AMODE == 0)
      asm volatile("s_wait_asynccnt 0" ::: "memory");
#endif
    __syncthreads();

    // ---- B fragments (4), cached in registers, reused by both M tiles ----
    v16bf bfr[4];
    #pragma unroll
    for (int ni = 0; ni < 4; ++ni)
      #pragma unroll
      for (int i = 0; i < 16; ++i) {
        // B 32x16 layout: lanes<16 K=0..15, lanes>=16 K=16..31
        bfr[ni][i] = Bs[wc + ni * 16 + lr][i + (hi << 4)];
      }
    #pragma unroll
    for (int mi = 0; mi < 2; ++mi) {
      v16bf af;
      #pragma unroll
      for (int i = 0; i < 16; ++i) {
        // A 16x32 layout (matches B): lanes<16 K=0..15, lanes>=16 K=16..31
        int kk = i + (hi << 4);
        af[i] = As[wr + mi * 16 + lr][kk];
      }
      #pragma unroll
      for (int ni = 0; ni < 4; ++ni)
        acc[mi][ni] = __builtin_amdgcn_wmma_f32_16x16x32_bf16(
            false, af, false, bfr[ni], (short)0, acc[mi][ni], false, false);
    }
    __syncthreads();
  }

  // ---- epilogue / store (C/D layout: vgpr e, lanes<16 row=e, >=16 row=8+e) ----
  #pragma unroll
  for (int mi = 0; mi < 2; ++mi)
    #pragma unroll
    for (int ni = 0; ni < 4; ++ni)
      #pragma unroll
      for (int e = 0; e < 8; ++e) {
        int gm = m0 + wr + mi * 16 + hi * 8 + e;
        int gn = n0 + wc + ni * 16 + lr;
        if (gn >= N) continue;
        float v = acc[mi][ni][e];
        size_t off = (size_t)gm * N + gn;
        if (EMODE == EP_F32)         ((float*)Cout)[off] = v;
        else if (EMODE == EP_ADDF32) ((float*)Cout)[off] += v;
        else if (EMODE == EP_BF16)   ((bf16*)Cout)[off] = f2bf(v);
        else if (EMODE == EP_TANH)   ((bf16*)Cout)[off] = f2bf(tanhf(v));
        else if (EMODE == EP_SIG)    ((bf16*)Cout)[off] = f2bf(sigm(v));
        else {  // EP_RELUSQ
          float r = v > 0.0f ? v : 0.0f;
          ((bf16*)Cout)[off] = f2bf(r * r);
        }
      }
}

// ---------------------------------------------------------------------------
// LayerNorm over C=1024, one row per block, bf16 output (feeds GEMM A).
// ---------------------------------------------------------------------------
__global__ __launch_bounds__(1024) void ln_bf16_kernel(
    const float* __restrict__ x, const float* __restrict__ g,
    const float* __restrict__ b, bf16* __restrict__ out)
{
  __shared__ float red[1024];
  int tid = threadIdx.x;
  size_t off = (size_t)blockIdx.x * RC + tid;
  float v = x[off];
  red[tid] = v; __syncthreads();
  for (int s = 512; s > 0; s >>= 1) { if (tid < s) red[tid] += red[tid + s]; __syncthreads(); }
  float mean = red[0] * (1.0f / RC);
  __syncthreads();
  float d = v - mean;
  red[tid] = d * d; __syncthreads();
  for (int s = 512; s > 0; s >>= 1) { if (tid < s) red[tid] += red[tid + s]; __syncthreads(); }
  float var = red[0] * (1.0f / RC);
  out[off] = f2bf(d * rsqrtf(var + 1e-5f) * g[tid] + b[tid]);
}

// ---------------------------------------------------------------------------
// Elementwise kernels
// ---------------------------------------------------------------------------
__global__ void cvt_bf16_kernel(const float* __restrict__ src, bf16* __restrict__ dst, size_t n) {
  for (size_t i = (size_t)blockIdx.x * blockDim.x + threadIdx.x; i < n;
       i += (size_t)gridDim.x * blockDim.x)
    dst[i] = f2bf(src[i]);
}

__global__ void copy_f32_kernel(const float* __restrict__ src, float* __restrict__ dst, size_t n) {
  for (size_t i = (size_t)blockIdx.x * blockDim.x + threadIdx.x; i < n;
       i += (size_t)gridDim.x * blockDim.x)
    dst[i] = src[i];
}

// decay = exp(-exp(-softplus(-(w0 + s)) - 0.5))
__global__ void decay_kernel(const float* __restrict__ s, const float* __restrict__ w0,
                             float* __restrict__ out, size_t n) {
  for (size_t i = (size_t)blockIdx.x * blockDim.x + threadIdx.x; i < n;
       i += (size_t)gridDim.x * blockDim.x) {
    int c = (int)(i & (RC - 1));
    float u = w0[c] + s[i];
    float w = -log1pf(expf(-u)) - 0.5f;
    out[i] = expf(-expf(w));
  }
}

// a = sigmoid(a0 + s)
__global__ void a_kernel(const float* __restrict__ s, const float* __restrict__ a0,
                         float* __restrict__ out, size_t n) {
  for (size_t i = (size_t)blockIdx.x * blockDim.x + threadIdx.x; i < n;
       i += (size_t)gridDim.x * blockDim.x) {
    int c = (int)(i & (RC - 1));
    out[i] = sigm(a0[c] + s[i]);
  }
}

// v = v + (v_first - v) * sigmoid(v0 + s)   (in-place on v)
__global__ void v_kernel(const float* __restrict__ s, const float* __restrict__ v0,
                         const float* __restrict__ vfirst, float* __restrict__ v, size_t n) {
  for (size_t i = (size_t)blockIdx.x * blockDim.x + threadIdx.x; i < n;
       i += (size_t)gridDim.x * blockDim.x) {
    int c = (int)(i & (RC - 1));
    float g = sigm(v0[c] + s[i]);
    float vv = v[i];
    v[i] = vv + (vfirst[i] - vv) * g;
  }
}

// ---------------------------------------------------------------------------
// RWKV-7 scan: one block per (b,h), 64 threads; thread i owns state row i
// (64 f32 registers). state = state*diag(d) + (state@a)⊗b + v⊗k ; y = state@r
// ---------------------------------------------------------------------------
__global__ __launch_bounds__(64) void scan_kernel(
    const float* __restrict__ r, const float* __restrict__ dcy,
    const float* __restrict__ k, const float* __restrict__ v,
    const float* __restrict__ a, const float* __restrict__ k_k,
    const float* __restrict__ k_a, float* __restrict__ y, int Tlen)
{
  const int bh = blockIdx.x;
  const int b  = bh / RH;
  const int h  = bh % RH;
  const int tid = threadIdx.x;   // state row i and staging lane j

  float S[RN];
  #pragma unroll
  for (int j = 0; j < RN; ++j) S[j] = 0.0f;

  __shared__ float sd[RN], sa[RN], sb[RN], sk[RN], sr[RN], red[RN];

  const float kkc = k_k[h * RN + tid];
  const float kac = k_a[h * RN + tid];
  const size_t base = (size_t)b * Tlen * RC + (size_t)h * RN;

  for (int t = 0; t < Tlen; ++t) {
    const size_t off = base + (size_t)t * RC;
    float kj = k[off + tid];
    float aj = a[off + tid];
    float kkj = kj * kkc;
    red[tid] = kkj * kkj;
    __syncthreads();
    for (int s = 32; s > 0; s >>= 1) { if (tid < s) red[tid] += red[tid + s]; __syncthreads(); }
    float inv = 1.0f / fmaxf(sqrtf(red[0]), 1e-12f);
    float kkn = kkj * inv;
    sd[tid] = dcy[off + tid];
    sa[tid] = -kkn;                             // a_t  = -kk
    sb[tid] = kkn * aj;                         // b_t  = kk * a
    sk[tid] = kj * (1.0f + (aj - 1.0f) * kac);  // k final
    sr[tid] = r[off + tid];
    __syncthreads();

    float vi = v[off + tid];
    float sad = 0.0f;
    #pragma unroll
    for (int j = 0; j < RN; ++j) sad += S[j] * sa[j];
    float yi = 0.0f;
    #pragma unroll
    for (int j = 0; j < RN; ++j) {
      S[j] = S[j] * sd[j] + sad * sb[j] + vi * sk[j];
      yi += S[j] * sr[j];
    }
    y[off + tid] = yi;
    __syncthreads();
  }
}

// ---------------------------------------------------------------------------
// Post: per-head GroupNorm(eps=6.4e-4) + ((r*k*r_k).sum)*v bonus + gate by g,
// write bf16 for the W_o GEMM. One block per (b,t) row, 1024 threads.
// ---------------------------------------------------------------------------
__global__ __launch_bounds__(1024) void post_kernel(
    const float* __restrict__ y, const float* __restrict__ r,
    const float* __restrict__ k, const float* __restrict__ v,
    const float* __restrict__ a, const float* __restrict__ g,
    const float* __restrict__ lnxg, const float* __restrict__ lnxb,
    const float* __restrict__ k_a, const float* __restrict__ r_k,
    bf16* __restrict__ out)
{
  __shared__ float red[1024];
  int tid = threadIdx.x;
  int hbase = tid & ~(RN - 1);
  size_t off = (size_t)blockIdx.x * RC + tid;

  float yv = y[off];
  red[tid] = yv; __syncthreads();
  for (int s = 32; s > 0; s >>= 1) { if ((tid & 63) < s) red[tid] += red[tid + s]; __syncthreads(); }
  float mean = red[hbase] * (1.0f / RN);
  __syncthreads();
  float d = yv - mean;
  red[tid] = d * d; __syncthreads();
  for (int s = 32; s > 0; s >>= 1) { if ((tid & 63) < s) red[tid] += red[tid + s]; __syncthreads(); }
  float var = red[hbase] * (1.0f / RN);
  __syncthreads();
  float yn = d * rsqrtf(var + 0.00064f) * lnxg[tid] + lnxb[tid];

  float rv = r[off];
  float kv = k[off];
  float av = a[off];
  float kf = kv * (1.0f + (av - 1.0f) * k_a[tid]);
  red[tid] = rv * kf * r_k[tid];   // r_k flat [H*N] == [C]
  __syncthreads();
  for (int s = 32; s > 0; s >>= 1) { if ((tid & 63) < s) red[tid] += red[tid + s]; __syncthreads(); }
  float dot = red[hbase];
  float y2 = yn + dot * v[off];
  out[off] = f2bf(y2 * g[off]);
}

// ---------------------------------------------------------------------------
// Host-side orchestration
// ---------------------------------------------------------------------------
extern "C" void kernel_launch(void* const* d_in, const int* in_sizes, int n_in,
                              void* d_out, int out_size, void* d_ws, size_t ws_size,
                              hipStream_t stream) {
  (void)in_sizes; (void)n_in; (void)out_size; (void)ws_size;

  // ---- inputs (setup_inputs dict order) ----
  const float* x       = (const float*)d_in[0];
  const float* v_first = (const float*)d_in[1];
  const float* ln1_g = (const float*)d_in[2];
  const float* ln1_b = (const float*)d_in[3];
  const float* ln2_g = (const float*)d_in[4];
  const float* ln2_b = (const float*)d_in[5];
  const float* x_r = (const float*)d_in[6];
  const float* x_w = (const float*)d_in[7];
  const float* x_k = (const float*)d_in[8];
  const float* x_v = (const float*)d_in[9];
  const float* x_a = (const float*)d_in[10];
  const float* x_g = (const float*)d_in[11];
  const float* w0  = (const float*)d_in[12];
  const float* w1  = (const float*)d_in[13];
  const float* w2  = (const float*)d_in[14];
  const float* a0  = (const float*)d_in[15];
  const float* a1  = (const float*)d_in[16];
  const float* a2  = (const float*)d_in[17];
  const float* v0  = (const float*)d_in[18];
  const float* v1  = (const float*)d_in[19];
  const float* v2  = (const float*)d_in[20];
  const float* g1  = (const float*)d_in[21];
  const float* g2  = (const float*)d_in[22];
  const float* k_k = (const float*)d_in[23];
  const float* k_a = (const float*)d_in[24];
  const float* r_k = (const float*)d_in[25];
  const float* W_r = (const float*)d_in[26];
  const float* W_k = (const float*)d_in[27];
  const float* W_v = (const float*)d_in[28];
  const float* W_o = (const float*)d_in[29];
  const float* lnx_g = (const float*)d_in[30];
  const float* lnx_b = (const float*)d_in[31];
  const float* cmix_x_k = (const float*)d_in[32];
  const float* W_key = (const float*)d_in[33];
  const float* W_val = (const float*)d_in[34];

  float* out_x  = (float*)d_out;        // [B,T,C]
  float* out_vf = out_x + BTC;          // [B,T,C] (v_first pass-through)

  // ---- workspace layout ----
  char* wsb = (char*)d_ws;
  size_t off = 0;
  auto alloc = [&](size_t bytes) -> char* {
    char* p = wsb + off;
    off = (off + bytes + 255) & ~(size_t)255;
    return p;
  };
  const size_t CC = (size_t)RC * RC;
  bf16* wr_bf  = (bf16*)alloc(CC * 2);
  bf16* wk_bf  = (bf16*)alloc(CC * 2);
  bf16* wv_bf  = (bf16*)alloc(CC * 2);
  bf16* wo_bf  = (bf16*)alloc(CC * 2);
  bf16* w1_bf  = (bf16*)alloc((size_t)RC * 64 * 2);
  bf16* w2_bf  = (bf16*)alloc((size_t)64 * RC * 2);
  bf16* a1_bf  = (bf16*)alloc((size_t)RC * 64 * 2);
  bf16* a2_bf  = (bf16*)alloc((size_t)64 * RC * 2);
  bf16* v1_bf  = (bf16*)alloc((size_t)RC * 32 * 2);
  bf16* v2_bf  = (bf16*)alloc((size_t)32 * RC * 2);
  bf16* g1_bf  = (bf16*)alloc((size_t)RC * 160 * 2);
  bf16* g2_bf  = (bf16*)alloc((size_t)160 * RC * 2);
  bf16* wkey_bf = (bf16*)alloc(CC * 4 * 2);
  bf16* wval_bf = (bf16*)alloc(CC * 4 * 2);

  bf16*  lnx_bf = (bf16*)alloc(BTC * 2);
  float* rbuf = (float*)alloc(BTC * 4);
  float* kbuf = (float*)alloc(BTC * 4);
  float* vbuf = (float*)alloc(BTC * 4);
  float* abuf = (float*)alloc(BTC * 4);
  float* dbuf = (float*)alloc(BTC * 4);
  float* gbuf = (float*)alloc(BTC * 4);
  float* tmp  = (float*)alloc(BTC * 4);
  bf16*  t1_bf = (bf16*)alloc((size_t)RM * 64 * 2);
  bf16*  t2_bf = (bf16*)alloc((size_t)RM * 64 * 2);
  bf16*  t3_bf = (bf16*)alloc((size_t)RM * 32 * 2);
  bf16*  t4_bf = (bf16*)alloc((size_t)RM * 160 * 2);
  float* ybuf  = (float*)alloc(BTC * 4);
  bf16*  yg_bf = (bf16*)alloc(BTC * 2);
  bf16*  ln2_bf = (bf16*)alloc(BTC * 2);
  bf16*  hid_bf = (bf16*)alloc((size_t)RM * 4096 * 2);

  auto cvt = [&](const float* src, bf16* dst, size_t n) {
    cvt_bf16_kernel<<<1024, 256, 0, stream>>>(src, dst, n);
  };
  auto gemm = [&](const bf16* A, const float* mix, const bf16* Bw, void* C,
                  int M, int N, int K, int amode, int emode) {
    dim3 grid((N + GBN - 1) / GBN, M / GBM);
#define GEMM_CASE(a, e)                                                        \
    if (amode == (a) && emode == (e)) {                                        \
      gemm_wmma_kernel<a, e><<<grid, 256, 0, stream>>>(A, mix, Bw, C, M, N, K);\
      return;                                                                  \
    }
    GEMM_CASE(0, EP_F32)
    GEMM_CASE(0, EP_ADDF32)
    GEMM_CASE(0, EP_BF16)
    GEMM_CASE(1, EP_F32)
    GEMM_CASE(1, EP_BF16)
    GEMM_CASE(1, EP_TANH)
    GEMM_CASE(1, EP_SIG)
    GEMM_CASE(1, EP_RELUSQ)
#undef GEMM_CASE
  };

  // ---- weight conversions f32 -> bf16 ----
  cvt(W_r, wr_bf, CC);  cvt(W_k, wk_bf, CC);  cvt(W_v, wv_bf, CC);  cvt(W_o, wo_bf, CC);
  cvt(w1, w1_bf, (size_t)RC * 64);  cvt(w2, w2_bf, (size_t)64 * RC);
  cvt(a1, a1_bf, (size_t)RC * 64);  cvt(a2, a2_bf, (size_t)64 * RC);
  cvt(v1, v1_bf, (size_t)RC * 32);  cvt(v2, v2_bf, (size_t)32 * RC);
  cvt(g1, g1_bf, (size_t)RC * 160); cvt(g2, g2_bf, (size_t)160 * RC);
  cvt(W_key, wkey_bf, CC * 4);      cvt(W_val, wval_bf, CC * 4);

  // ---- outputs seeded: out_x = x (residual base), out_vf = v_first ----
  copy_f32_kernel<<<2048, 256, 0, stream>>>(x, out_x, BTC);
  copy_f32_kernel<<<2048, 256, 0, stream>>>(v_first, out_vf, BTC);

  // ---- ln1 ----
  ln_bf16_kernel<<<RM, 1024, 0, stream>>>(x, ln1_g, ln1_b, lnx_bf);

  // ---- big projections (token-shift mix fused in A load) ----
  gemm(lnx_bf, x_r, wr_bf, rbuf, RM, RC, RC, 1, EP_F32);
  gemm(lnx_bf, x_k, wk_bf, kbuf, RM, RC, RC, 1, EP_F32);
  gemm(lnx_bf, x_v, wv_bf, vbuf, RM, RC, RC, 1, EP_F32);

  // ---- decay LoRA: tanh(xw@w1)@w2, then elementwise decay ----
  gemm(lnx_bf, x_w, w1_bf, t1_bf, RM, 64, RC, 1, EP_TANH);
  gemm(t1_bf, nullptr, w2_bf, tmp, RM, RC, 64, 0, EP_F32);
  decay_kernel<<<2048, 256, 0, stream>>>(tmp, w0, dbuf, BTC);

  // ---- a LoRA: sigmoid(a0 + (xa@a1)@a2) ----
  gemm(lnx_bf, x_a, a1_bf, t2_bf, RM, 64, RC, 1, EP_BF16);
  gemm(t2_bf, nullptr, a2_bf, tmp, RM, RC, 64, 0, EP_F32);
  a_kernel<<<2048, 256, 0, stream>>>(tmp, a0, abuf, BTC);

  // ---- v mix with v_first: v += (v_first - v)*sigmoid(v0 + (xv@v1)@v2) ----
  gemm(lnx_bf, x_v, v1_bf, t3_bf, RM, 32, RC, 1, EP_BF16);
  gemm(t3_bf, nullptr, v2_bf, tmp, RM, RC, 32, 0, EP_F32);
  v_kernel<<<2048, 256, 0, stream>>>(tmp, v0, v_first, vbuf, BTC);

  // ---- gate: sigmoid(xg@g1)@g2 ----
  gemm(lnx_bf, x_g, g1_bf, t4_bf, RM, 160, RC, 1, EP_SIG);
  gemm(t4_bf, nullptr, g2_bf, gbuf, RM, RC, 160, 0, EP_F32);

  // ---- sequential RWKV-7 state scan (64 independent (b,h) recurrences) ----
  scan_kernel<<<RB * RH, RN, 0, stream>>>(rbuf, dbuf, kbuf, vbuf, abuf, k_k, k_a, ybuf, RT);

  // ---- groupnorm + r*k*r_k bonus + gating -> bf16 ----
  post_kernel<<<RM, 1024, 0, stream>>>(ybuf, rbuf, kbuf, vbuf, abuf, gbuf,
                                       lnx_g, lnx_b, k_a, r_k, yg_bf);

  // ---- attention output projection, accumulated into residual ----
  gemm(yg_bf, nullptr, wo_bf, out_x, RM, RC, RC, 0, EP_ADDF32);

  // ---- cmix: ln2 -> shift-mix -> relu^2(.@W_key) -> @W_val accumulate ----
  ln_bf16_kernel<<<RM, 1024, 0, stream>>>(out_x, ln2_g, ln2_b, ln2_bf);
  gemm(ln2_bf, cmix_x_k, wkey_bf, hid_bf, RM, 4 * RC, RC, 1, EP_RELUSQ);
  gemm(hid_bf, nullptr, wval_bf, out_x, RM, RC, 4 * RC, 0, EP_ADDF32);
}
